// HEPOSBartSummarizer_21406117003654
// MI455X (gfx1250) — matrix operand order repaired and checked by reading, not compiled, not run on hardware
//
#include <hip/hip_runtime.h>
#include <hip/hip_bf16.h>

// ---------------- model constants ----------------
#define DMODEL 768
#define NHEAD  12
#define DHEAD  64
#define DFF    3072
#define NLE    2
#define NLD    2
#define SLEN   2048
#define TLEN   512
#define NB     2
#define HSTRIDE 4
#define SKLEN  (SLEN / HSTRIDE)   // 512
#define VOCAB  50265
#define BOSID  2

typedef long long ll;
typedef __attribute__((ext_vector_type(16))) __bf16 v16bf;
typedef __attribute__((ext_vector_type(8)))  float  v8f;
typedef __attribute__((ext_vector_type(2)))  __bf16 bf16x2;
typedef __attribute__((ext_vector_type(2)))  float  f32x2;

union F16Frag { unsigned u[8]; v16bf v; };

// pack two f32 -> one dword of packed bf16 (single v_cvt_pk_bf16_f32)
__device__ __forceinline__ unsigned pk(float lo, float hi)
{
    f32x2 f; f[0] = lo; f[1] = hi;
    bf16x2 h = __builtin_convertvector(f, bf16x2);
    union { bf16x2 h; unsigned u; } t;
    t.h = h;
    return t.u;
}

// =====================================================================
// Strided-batched GEMM  C[z] = A[z] @ B[z] (+ bias)
// 256 threads = 8 waves; block tile 64(M) x 64(N); K step 32.
// Requires: M % 64 == 0, K % 32 == 0 (true for every call site).
// N ragged only on the LM-head edge tiles -> block-uniform slow path.
// LDS holds bf16 PAIRS (dwords): lsA[m][k>>1], lsB[n][k>>1] so each
// lane's WMMA fragment is two contiguous uint4 (ds_load_b128).
// TRANSB (compile-time): B stored [N,K] row-major.
// =====================================================================
template <bool TRANSB>
__global__ __launch_bounds__(256)
void gemm_wmma_bf16(const float* __restrict__ A, const float* __restrict__ Bm,
                    const float* __restrict__ bias, float* __restrict__ C,
                    int M, int N, int K, int lda, int ldb, int ldc,
                    ll sAb, ll sAh, ll sBb, ll sBh, ll sCb, ll sCh,
                    int Hdiv)
{
    __shared__ unsigned lsA[64][20];   // 64 rows x 16 pairs (+4 pad)
    __shared__ unsigned lsB[64][20];   // 64 cols x 16 pairs (+4 pad)

    const int z = blockIdx.z;
    A  += (ll)(z / Hdiv) * sAb + (ll)(z % Hdiv) * sAh;
    Bm += (ll)(z / Hdiv) * sBb + (ll)(z % Hdiv) * sBh;
    C  += (ll)(z / Hdiv) * sCb + (ll)(z % Hdiv) * sCh;

    const int m0   = blockIdx.y * 64;
    const int n0   = blockIdx.x * 64;
    const int tid  = threadIdx.x;
    const int wave = tid >> 5, lane = tid & 31;
    const int half = lane >> 4, l15 = lane & 15;
    const int mi   = wave & 3,  ni  = wave >> 2;

    // staging decomposition
    const int r0  = tid >> 3;        // 0..31 : row (A) / col (B-trans)
    const int kq  = (tid & 7) * 4;   // 0,4,...,28 : k quad
    const int kp0 = tid >> 5;        // 0..7  : k-pair (B non-trans)
    const int np  = tid & 31;        // 0..31 : n-pair (B non-trans)

    v8f acc0 = {0.f,0.f,0.f,0.f,0.f,0.f,0.f,0.f};
    v8f acc1 = {0.f,0.f,0.f,0.f,0.f,0.f,0.f,0.f};

    const int  mrow = mi * 16 + l15;
    const int  nrow = ni * 32 + l15;
    const bool fullN = (n0 + 64 <= N);

    if (fullN) {
        float4 a0, a1, b0, b1;
        float2 g00, g01, g10, g11;
        // ---- prologue: load K-step 0 ----
        {
            const float* Ab = A + (ll)(m0 + r0) * lda + kq;
            a0 = *(const float4*)Ab;
            a1 = *(const float4*)(Ab + (ll)32 * lda);
            if constexpr (TRANSB) {
                const float* Bb = Bm + (ll)(n0 + r0) * ldb + kq;
                b0 = *(const float4*)Bb;
                b1 = *(const float4*)(Bb + (ll)32 * ldb);
            } else {
                const float* Bb = Bm + (ll)(2 * kp0) * ldb + n0 + 2 * np;
                g00 = *(const float2*)Bb;
                g01 = *(const float2*)(Bb + ldb);
                g10 = *(const float2*)(Bb + (ll)16 * ldb);
                g11 = *(const float2*)(Bb + (ll)17 * ldb);
            }
        }
        for (int k0 = 0; k0 < K; k0 += 32) {
            // ---- stage regs -> LDS (packed bf16 pairs) ----
            {
                uint2 t0; t0.x = pk(a0.x, a0.y); t0.y = pk(a0.z, a0.w);
                *(uint2*)&lsA[r0][kq >> 1] = t0;
                uint2 t1; t1.x = pk(a1.x, a1.y); t1.y = pk(a1.z, a1.w);
                *(uint2*)&lsA[r0 + 32][kq >> 1] = t1;
            }
            if constexpr (TRANSB) {
                uint2 t0; t0.x = pk(b0.x, b0.y); t0.y = pk(b0.z, b0.w);
                *(uint2*)&lsB[r0][kq >> 1] = t0;
                uint2 t1; t1.x = pk(b1.x, b1.y); t1.y = pk(b1.z, b1.w);
                *(uint2*)&lsB[r0 + 32][kq >> 1] = t1;
            } else {
                lsB[2 * np    ][kp0    ] = pk(g00.x, g01.x);
                lsB[2 * np + 1][kp0    ] = pk(g00.y, g01.y);
                lsB[2 * np    ][kp0 + 8] = pk(g10.x, g11.x);
                lsB[2 * np + 1][kp0 + 8] = pk(g10.y, g11.y);
            }
            __syncthreads();
            // ---- issue next K-step's global loads (overlap with WMMA) ----
            if (k0 + 32 < K) {
                const float* Ab = A + (ll)(m0 + r0) * lda + (k0 + 32) + kq;
                a0 = *(const float4*)Ab;
                a1 = *(const float4*)(Ab + (ll)32 * lda);
                if constexpr (TRANSB) {
                    const float* Bb = Bm + (ll)(n0 + r0) * ldb + (k0 + 32) + kq;
                    b0 = *(const float4*)Bb;
                    b1 = *(const float4*)(Bb + (ll)32 * ldb);
                } else {
                    const float* Bb = Bm + (ll)(k0 + 32 + 2 * kp0) * ldb + n0 + 2 * np;
                    g00 = *(const float2*)Bb;
                    g01 = *(const float2*)(Bb + ldb);
                    g10 = *(const float2*)(Bb + (ll)16 * ldb);
                    g11 = *(const float2*)(Bb + (ll)17 * ldb);
                }
            }
            // ---- fragments: two ds_load_b128 each ----
            F16Frag fa, fb0, fb1;
            *(uint4*)&fa.u[0]  = *(const uint4*)&lsA[mrow][half * 8];
            *(uint4*)&fa.u[4]  = *(const uint4*)&lsA[mrow][half * 8 + 4];
            *(uint4*)&fb0.u[0] = *(const uint4*)&lsB[nrow][half * 8];
            *(uint4*)&fb0.u[4] = *(const uint4*)&lsB[nrow][half * 8 + 4];
            *(uint4*)&fb1.u[0] = *(const uint4*)&lsB[nrow + 16][half * 8];
            *(uint4*)&fb1.u[4] = *(const uint4*)&lsB[nrow + 16][half * 8 + 4];
            acc0 = __builtin_amdgcn_wmma_f32_16x16x32_bf16(false, fa.v, false, fb0.v,
                                                           (short)0, acc0, false, false);
            acc1 = __builtin_amdgcn_wmma_f32_16x16x32_bf16(false, fa.v, false, fb1.v,
                                                           (short)0, acc1, false, false);
            __syncthreads();
        }
    } else {
        // ---- N-edge (LM head last tile column): guarded scalar staging ----
        for (int k0 = 0; k0 < K; k0 += 32) {
            {
                const float* Ab = A + (ll)(m0 + r0) * lda + k0 + kq;
                float4 a0 = *(const float4*)Ab;
                float4 a1 = *(const float4*)(Ab + (ll)32 * lda);
                uint2 t0; t0.x = pk(a0.x, a0.y); t0.y = pk(a0.z, a0.w);
                *(uint2*)&lsA[r0][kq >> 1] = t0;
                uint2 t1; t1.x = pk(a1.x, a1.y); t1.y = pk(a1.z, a1.w);
                *(uint2*)&lsA[r0 + 32][kq >> 1] = t1;
            }
            for (int e = tid; e < 64 * 16; e += 256) {
                int n = e >> 4, kp = e & 15;
                int gn = n0 + n, gk = k0 + 2 * kp;
                float f0 = 0.f, f1 = 0.f;
                if (gn < N) {
                    if constexpr (TRANSB) {
                        f0 = Bm[(ll)gn * ldb + gk];
                        f1 = Bm[(ll)gn * ldb + gk + 1];
                    } else {
                        f0 = Bm[(ll)gk * ldb + gn];
                        f1 = Bm[(ll)(gk + 1) * ldb + gn];
                    }
                }
                lsB[n][kp] = pk(f0, f1);
            }
            __syncthreads();
            F16Frag fa, fb0, fb1;
            *(uint4*)&fa.u[0]  = *(const uint4*)&lsA[mrow][half * 8];
            *(uint4*)&fa.u[4]  = *(const uint4*)&lsA[mrow][half * 8 + 4];
            *(uint4*)&fb0.u[0] = *(const uint4*)&lsB[nrow][half * 8];
            *(uint4*)&fb0.u[4] = *(const uint4*)&lsB[nrow][half * 8 + 4];
            *(uint4*)&fb1.u[0] = *(const uint4*)&lsB[nrow + 16][half * 8];
            *(uint4*)&fb1.u[4] = *(const uint4*)&lsB[nrow + 16][half * 8 + 4];
            acc0 = __builtin_amdgcn_wmma_f32_16x16x32_bf16(false, fa.v, false, fb0.v,
                                                           (short)0, acc0, false, false);
            acc1 = __builtin_amdgcn_wmma_f32_16x16x32_bf16(false, fa.v, false, fb1.v,
                                                           (short)0, acc1, false, false);
            __syncthreads();
        }
    }

    // ---- store D (+bias): VGPR j -> M = j + 8*half ----
    const int gnA = n0 + nrow;
    const int gnB = gnA + 16;
    const float bvA = (bias && gnA < N) ? bias[gnA] : 0.f;
    const float bvB = (bias && gnB < N) ? bias[gnB] : 0.f;
#pragma unroll
    for (int j = 0; j < 8; ++j) {
        const int gm = m0 + mi * 16 + half * 8 + j;   // always < M
        if (gnA < N) C[(ll)gm * ldc + gnA] = acc0[j] + bvA;
        if (gnB < N) C[(ll)gm * ldc + gnB] = acc1[j] + bvB;
    }
}

// ---------------- block reduction helper (256 threads) ----------------
__device__ __forceinline__ float blkSum(float v, float* sbuf)
{
    int tid = threadIdx.x;
    sbuf[tid] = v; __syncthreads();
    for (int s = 128; s > 0; s >>= 1) {
        if (tid < s) sbuf[tid] += sbuf[tid + s];
        __syncthreads();
    }
    float r = sbuf[0]; __syncthreads();
    return r;
}

// ---------------- embedding + LayerNorm (enc direct; dec shift+BOS) -------
__global__ __launch_bounds__(256)
void embed_ln(const int* __restrict__ ids, const float* __restrict__ emb,
              const float* __restrict__ pos, const float* __restrict__ lns,
              const float* __restrict__ lnb, float* __restrict__ out,
              int seqlen, int dec)
{
    __shared__ float sbuf[256];
    const int row = blockIdx.x;
    const int t   = row % seqlen;
    int id;
    if (dec) id = (t == 0) ? BOSID : ids[row - 1];
    else     id = ids[row];
    const int tid = threadIdx.x;
    float v[3];
#pragma unroll
    for (int u = 0; u < 3; ++u) {
        int d = tid + u * 256;
        v[u] = emb[(ll)id * DMODEL + d] + pos[(ll)t * DMODEL + d];
    }
    float mean = blkSum(v[0] + v[1] + v[2], sbuf) * (1.0f / DMODEL);
    float q = 0.f;
#pragma unroll
    for (int u = 0; u < 3; ++u) { float dv = v[u] - mean; q += dv * dv; }
    float var  = blkSum(q, sbuf) * (1.0f / DMODEL);
    float rstd = rsqrtf(var + 1e-5f);
#pragma unroll
    for (int u = 0; u < 3; ++u) {
        int d = tid + u * 256;
        out[(ll)row * DMODEL + d] = (v[u] - mean) * rstd * lns[d] + lnb[d];
    }
}

// ---------------- residual add + LayerNorm --------------------------------
__global__ __launch_bounds__(256)
void add_ln(const float* __restrict__ x, const float* __restrict__ r,
            const float* __restrict__ lns, const float* __restrict__ lnb,
            float* __restrict__ out)
{
    __shared__ float sbuf[256];
    const int row = blockIdx.x;
    const int tid = threadIdx.x;
    float v[3];
#pragma unroll
    for (int u = 0; u < 3; ++u) {
        int d = tid + u * 256;
        v[u] = x[(ll)row * DMODEL + d] + r[(ll)row * DMODEL + d];
    }
    float mean = blkSum(v[0] + v[1] + v[2], sbuf) * (1.0f / DMODEL);
    float q = 0.f;
#pragma unroll
    for (int u = 0; u < 3; ++u) { float dv = v[u] - mean; q += dv * dv; }
    float var  = blkSum(q, sbuf) * (1.0f / DMODEL);
    float rstd = rsqrtf(var + 1e-5f);
#pragma unroll
    for (int u = 0; u < 3; ++u) {
        int d = tid + u * 256;
        out[(ll)row * DMODEL + d] = (v[u] - mean) * rstd * lns[d] + lnb[d];
    }
}

// ---------------- scaled (optionally causal) row softmax ------------------
__global__ __launch_bounds__(256)
void softmax_rows(float* __restrict__ Sc, int ncols, float scale,
                  int causal, int qmod)
{
    __shared__ float sbuf[256];
    const int row = blockIdx.x;
    float* p = Sc + (ll)row * ncols;
    const int limit = causal ? (row % qmod) + 1 : ncols;
    const int tid = threadIdx.x;

    float mx = -3.0e38f;
    for (int c = tid; c < limit; c += 256) mx = fmaxf(mx, p[c] * scale);
    sbuf[tid] = mx; __syncthreads();
    for (int s = 128; s > 0; s >>= 1) {
        if (tid < s) sbuf[tid] = fmaxf(sbuf[tid], sbuf[tid + s]);
        __syncthreads();
    }
    mx = sbuf[0]; __syncthreads();

    float sum = 0.f;
    for (int c = tid; c < limit; c += 256) {
        float e = expf(p[c] * scale - mx);
        p[c] = e; sum += e;
    }
    sum = blkSum(sum, sbuf);
    float inv = 1.0f / sum;
    for (int c = tid; c < limit; c += 256) p[c] *= inv;
    for (int c = limit + tid; c < ncols; c += 256) p[c] = 0.f;
}

// ---------------- tanh-approx GELU ----------------------------------------
__global__ void gelu_k(float* __restrict__ x, ll n)
{
    ll i = (ll)blockIdx.x * blockDim.x + threadIdx.x;
    ll stride = (ll)gridDim.x * blockDim.x;
    for (; i < n; i += stride) {
        float v = x[i];
        float c = 0.7978845608028654f * (v + 0.044715f * v * v * v);
        x[i] = 0.5f * v * (1.0f + tanhf(c));
    }
}

// ---------------- HEPOS gather: [B,S,D] -> [B,H,SK,DH], float4 ------------
__global__ void hepos_gather(const float* __restrict__ src, float* __restrict__ dst)
{
    int i = blockIdx.x * blockDim.x + threadIdx.x;
    const int total4 = NB * NHEAD * SKLEN * DHEAD / 4;
    if (i >= total4) return;
    int d4   = i & 15;                  // 16 float4 per 64-wide row
    int rest = i >> 4;                  // b*H*SK + h*SK + j
    int j = rest % SKLEN;
    int h = (rest / SKLEN) % NHEAD;
    int b = rest / (SKLEN * NHEAD);
    int spos = (h & (HSTRIDE - 1)) + HSTRIDE * j;
    *(float4*)&dst[(ll)i * 4] =
        *(const float4*)&src[((ll)(b * SLEN + spos)) * DMODEL + h * DHEAD + d4 * 4];
}

// ---------------- per-row NLL: lse - logit[label] -------------------------
__global__ __launch_bounds__(256)
void row_loss(const float* __restrict__ logits, const int* __restrict__ labels,
              float* __restrict__ rl)
{
    __shared__ float sbuf[256];
    const int row = blockIdx.x;
    const float* p = logits + (ll)row * VOCAB;
    const int tid = threadIdx.x;

    float mx = -3.0e38f;
    for (int c = tid; c < VOCAB; c += 256) mx = fmaxf(mx, p[c]);
    sbuf[tid] = mx; __syncthreads();
    for (int s = 128; s > 0; s >>= 1) {
        if (tid < s) sbuf[tid] = fmaxf(sbuf[tid], sbuf[tid + s]);
        __syncthreads();
    }
    mx = sbuf[0]; __syncthreads();

    float sum = 0.f;
    for (int c = tid; c < VOCAB; c += 256) sum += expf(p[c] - mx);
    sum = blkSum(sum, sbuf);
    if (tid == 0) rl[row] = (mx + logf(sum)) - p[labels[row]];
}

__global__ __launch_bounds__(256)
void loss_reduce(const float* __restrict__ rl, int n, float* __restrict__ out)
{
    __shared__ float sbuf[256];
    float s = 0.f;
    for (int i = threadIdx.x; i < n; i += 256) s += rl[i];
    s = blkSum(s, sbuf);
    if (threadIdx.x == 0) *out = s / (float)n;
}

// =====================================================================
// Host-side launch helper  (M must be a multiple of 64 — true everywhere)
// =====================================================================
static inline void gemmL(hipStream_t st, const float* A, const float* B,
                         const float* bias, float* C,
                         int M, int N, int K, int lda, int ldb, int ldc,
                         ll sAb, ll sAh, ll sBb, ll sBh, ll sCb, ll sCh,
                         int Hdiv, int transB, int nz)
{
    dim3 g((N + 63) / 64, M / 64, nz);
    if (transB)
        gemm_wmma_bf16<true><<<g, 256, 0, st>>>(A, B, bias, C, M, N, K,
                                                lda, ldb, ldc,
                                                sAb, sAh, sBb, sBh, sCb, sCh, Hdiv);
    else
        gemm_wmma_bf16<false><<<g, 256, 0, st>>>(A, B, bias, C, M, N, K,
                                                 lda, ldb, ldc,
                                                 sAb, sAh, sBb, sBh, sCb, sCh, Hdiv);
}

extern "C" void kernel_launch(void* const* d_in, const int* in_sizes, int n_in,
                              void* d_out, int out_size, void* d_ws, size_t ws_size,
                              hipStream_t stream)
{
    (void)in_sizes; (void)n_in;
    // ---- inputs (setup_inputs dict order, nested dicts flattened) ----
    const int*   input_ids = (const int*)  d_in[0];
    const int*   labels    = (const int*)  d_in[1];
    const float* emb       = (const float*)d_in[2];
    const float* pos       = (const float*)d_in[3];
    const float* ln_s      = (const float*)d_in[4];
    const float* ln_b      = (const float*)d_in[5];
    // encoder
    const float* eWq=(const float*)d_in[6],  *ebq=(const float*)d_in[7];
    const float* eWk=(const float*)d_in[8],  *ebk=(const float*)d_in[9];
    const float* eWv=(const float*)d_in[10], *ebv=(const float*)d_in[11];
    const float* eWo=(const float*)d_in[12], *ebo=(const float*)d_in[13];
    const float* eln1s=(const float*)d_in[14], *eln1b=(const float*)d_in[15];
    const float* eW1=(const float*)d_in[16], *eb1=(const float*)d_in[17];
    const float* eW2=(const float*)d_in[18], *eb2=(const float*)d_in[19];
    const float* eln2s=(const float*)d_in[20], *eln2b=(const float*)d_in[21];
    // decoder
    const float* sWq=(const float*)d_in[22], *sbq=(const float*)d_in[23];
    const float* sWk=(const float*)d_in[24], *sbk=(const float*)d_in[25];
    const float* sWv=(const float*)d_in[26], *sbv=(const float*)d_in[27];
    const float* sWo=(const float*)d_in[28], *sbo=(const float*)d_in[29];
    const float* cWq=(const float*)d_in[30], *cbq=(const float*)d_in[31];
    const float* cWk=(const float*)d_in[32], *cbk=(const float*)d_in[33];
    const float* cWv=(const float*)d_in[34], *cbv=(const float*)d_in[35];
    const float* cWo=(const float*)d_in[36], *cbo=(const float*)d_in[37];
    const float* dln1s=(const float*)d_in[38], *dln1b=(const float*)d_in[39];
    const float* dln2s=(const float*)d_in[40], *dln2b=(const float*)d_in[41];
    const float* dW1=(const float*)d_in[42], *db1=(const float*)d_in[43];
    const float* dW2=(const float*)d_in[44], *db2=(const float*)d_in[45];
    const float* dln3s=(const float*)d_in[46], *dln3b=(const float*)d_in[47];

    const int ME = NB * SLEN;   // 4096 encoder tokens
    const int MD = NB * TLEN;   // 1024 decoder tokens
    const int D  = DMODEL;

    // ---- workspace layout (floats) ----
    float* W = (float*)d_ws;
    ll o = 0;
    float* X  = W + o; o += (ll)ME * D;
    float* R1 = W + o; o += (ll)ME * D;
    float* Q  = W + o; o += (ll)ME * D;
    float* Kb = W + o; o += (ll)ME * D;
    float* Vb = W + o; o += (ll)ME * D;
    float* AO = W + o; o += (ll)ME * D;
    float* FF = W + o; o += (ll)ME * DFF;
    float* SC = W + o; o += (ll)NB * NHEAD * TLEN * TLEN;  // >= S*S
    float* KH = W + o; o += (ll)NB * NHEAD * SKLEN * DHEAD;
    float* VH = W + o; o += (ll)NB * NHEAD * SKLEN * DHEAD;
    float* Y  = W + o; o += (ll)MD * D;
    float* RL = W + o; o += (ll)MD;
    if (ws_size < (size_t)o * sizeof(float)) return;

    // ================= encoder =================
    embed_ln<<<ME, 256, 0, stream>>>(input_ids, emb, pos, ln_s, ln_b, X, SLEN, 0);
    for (int i = 0; i < NLE; ++i) {
        const ll wDD = (ll)i * D * D, wD = (ll)i * D;
        gemmL(stream, X, eWq + wDD, ebq + wD, Q,  ME, D, D, D, D, D, 0,0,0,0,0,0, 1, 0, 1);
        gemmL(stream, X, eWk + wDD, ebk + wD, Kb, ME, D, D, D, D, D, 0,0,0,0,0,0, 1, 0, 1);
        gemmL(stream, X, eWv + wDD, ebv + wD, Vb, ME, D, D, D, D, D, 0,0,0,0,0,0, 1, 0, 1);
        // self-attention, one (b,h) at a time to bound score memory
        for (int zz = 0; zz < NB * NHEAD; ++zz) {
            int b = zz / NHEAD, h = zz % NHEAD;
            const float* Qp = Q  + (ll)b * SLEN * D + h * DHEAD;
            const float* Kp = Kb + (ll)b * SLEN * D + h * DHEAD;
            const float* Vp = Vb + (ll)b * SLEN * D + h * DHEAD;
            float*       Op = AO + (ll)b * SLEN * D + h * DHEAD;
            gemmL(stream, Qp, Kp, nullptr, SC, SLEN, SLEN, DHEAD, D, D, SLEN,
                  0,0,0,0,0,0, 1, 1, 1);
            softmax_rows<<<SLEN, 256, 0, stream>>>(SC, SLEN, 0.125f, 0, SLEN);
            gemmL(stream, SC, Vp, nullptr, Op, SLEN, DHEAD, SLEN, SLEN, D, D,
                  0,0,0,0,0,0, 1, 0, 1);
        }
        gemmL(stream, AO, eWo + wDD, ebo + wD, R1, ME, D, D, D, D, D, 0,0,0,0,0,0, 1, 0, 1);
        add_ln<<<ME, 256, 0, stream>>>(X, R1, eln1s + wD, eln1b + wD, X);
        // FFN
        gemmL(stream, X, eW1 + (ll)i * D * DFF, eb1 + (ll)i * DFF, FF,
              ME, DFF, D, D, DFF, DFF, 0,0,0,0,0,0, 1, 0, 1);
        gelu_k<<<4096, 256, 0, stream>>>(FF, (ll)ME * DFF);
        gemmL(stream, FF, eW2 + (ll)i * DFF * D, eb2 + wD, R1,
              ME, D, DFF, DFF, D, D, 0,0,0,0,0,0, 1, 0, 1);
        add_ln<<<ME, 256, 0, stream>>>(X, R1, eln2s + wD, eln2b + wD, X);
    }

    // ================= decoder =================
    embed_ln<<<MD, 256, 0, stream>>>(labels, emb, pos, ln_s, ln_b, Y, TLEN, 1);
    for (int i = 0; i < NLD; ++i) {
        const ll wDD = (ll)i * D * D, wD = (ll)i * D;
        // ---- causal self-attention (batched over all b,h) ----
        gemmL(stream, Y, sWq + wDD, sbq + wD, Q,  MD, D, D, D, D, D, 0,0,0,0,0,0, 1, 0, 1);
        gemmL(stream, Y, sWk + wDD, sbk + wD, Kb, MD, D, D, D, D, D, 0,0,0,0,0,0, 1, 0, 1);
        gemmL(stream, Y, sWv + wDD, sbv + wD, Vb, MD, D, D, D, D, D, 0,0,0,0,0,0, 1, 0, 1);
        gemmL(stream, Q, Kb, nullptr, SC, TLEN, TLEN, DHEAD, D, D, TLEN,
              (ll)TLEN * D, DHEAD, (ll)TLEN * D, DHEAD,
              (ll)NHEAD * TLEN * TLEN, (ll)TLEN * TLEN, NHEAD, 1, NB * NHEAD);
        softmax_rows<<<NB * NHEAD * TLEN, 256, 0, stream>>>(SC, TLEN, 0.125f, 1, TLEN);
        gemmL(stream, SC, Vb, nullptr, AO, TLEN, DHEAD, TLEN, TLEN, D, D,
              (ll)NHEAD * TLEN * TLEN, (ll)TLEN * TLEN, (ll)TLEN * D, DHEAD,
              (ll)TLEN * D, DHEAD, NHEAD, 0, NB * NHEAD);
        gemmL(stream, AO, sWo + wDD, sbo + wD, R1, MD, D, D, D, D, D, 0,0,0,0,0,0, 1, 0, 1);
        add_ln<<<MD, 256, 0, stream>>>(Y, R1, dln1s + wD, dln1b + wD, Y);
        // ---- HEPOS cross-attention ----
        gemmL(stream, Y, cWq + wDD, cbq + wD, Q,  MD, D, D, D, D, D, 0,0,0,0,0,0, 1, 0, 1);
        gemmL(stream, X, cWk + wDD, cbk + wD, Kb, ME, D, D, D, D, D, 0,0,0,0,0,0, 1, 0, 1);
        gemmL(stream, X, cWv + wDD, cbv + wD, Vb, ME, D, D, D, D, D, 0,0,0,0,0,0, 1, 0, 1);
        const int ng4 = NB * NHEAD * SKLEN * DHEAD / 4;
        hepos_gather<<<(ng4 + 255) / 256, 256, 0, stream>>>(Kb, KH);
        hepos_gather<<<(ng4 + 255) / 256, 256, 0, stream>>>(Vb, VH);
        gemmL(stream, Q, KH, nullptr, SC, TLEN, SKLEN, DHEAD, D, DHEAD, SKLEN,
              (ll)TLEN * D, DHEAD, (ll)NHEAD * SKLEN * DHEAD, (ll)SKLEN * DHEAD,
              (ll)NHEAD * TLEN * SKLEN, (ll)TLEN * SKLEN, NHEAD, 1, NB * NHEAD);
        softmax_rows<<<NB * NHEAD * TLEN, 256, 0, stream>>>(SC, SKLEN, 0.125f, 0, TLEN);
        gemmL(stream, SC, VH, nullptr, AO, TLEN, DHEAD, SKLEN, SKLEN, DHEAD, D,
              (ll)NHEAD * TLEN * SKLEN, (ll)TLEN * SKLEN,
              (ll)NHEAD * SKLEN * DHEAD, (ll)SKLEN * DHEAD,
              (ll)TLEN * D, DHEAD, NHEAD, 0, NB * NHEAD);
        gemmL(stream, AO, cWo + wDD, cbo + wD, R1, MD, D, D, D, D, D, 0,0,0,0,0,0, 1, 0, 1);
        add_ln<<<MD, 256, 0, stream>>>(Y, R1, dln2s + wD, dln2b + wD, Y);
        // ---- FFN ----
        gemmL(stream, Y, dW1 + (ll)i * D * DFF, db1 + (ll)i * DFF, FF,
              MD, DFF, D, D, DFF, DFF, 0,0,0,0,0,0, 1, 0, 1);
        gelu_k<<<2048, 256, 0, stream>>>(FF, (ll)MD * DFF);
        gemmL(stream, FF, dW2 + (ll)i * DFF * D, db2 + wD, R1,
              MD, D, DFF, DFF, D, D, 0,0,0,0,0,0, 1, 0, 1);
        add_ln<<<MD, 256, 0, stream>>>(Y, R1, dln3s + wD, dln3b + wD, Y);
    }

    // ================= tied LM head + loss =================
    float* logits = (float*)d_out;
    gemmL(stream, Y, emb, nullptr, logits, MD, VOCAB, D, D, D, VOCAB,
          0,0,0,0,0,0, 1, 1, 1);                       // emb [V,D], transB
    row_loss<<<MD, 256, 0, stream>>>(logits, labels, RL);
    loss_reduce<<<1, 256, 0, stream>>>(RL, MD, logits + ((ll)out_size - 1));
}
